// TensorBase_83253646065681
// MI455X (gfx1250) — matrix operand order repaired and checked by reading, not compiled, not run on hardware
//
#include <hip/hip_runtime.h>
#include <math.h>

typedef _Float16 f16;
typedef _Float16 v8h  __attribute__((ext_vector_type(8)));
typedef _Float16 v16h __attribute__((ext_vector_type(16)));
typedef float    v8f  __attribute__((ext_vector_type(8)));

#define NSAMP   256
#define TM      128          // points per block
#define FSTR    424          // feature row stride in halves (416 + 8 pad)
#define H1STR   136          // hidden row stride in halves (128 + 8 pad)
#define K1TILES 13           // 416 / 32
#define K2TILES 4            // 128 / 32
#define CHSTRIDE (128*128*128)

static __device__ __forceinline__ v16h cat8(v8h lo, v8h hi) {
  return __builtin_shufflevector(lo, hi, 0,1,2,3,4,5,6,7,8,9,10,11,12,13,14,15);
}

// ---------------------------------------------------------------------------
// Kernel 0: repack W1 (390x128, K padded to 416) and W2 (128x128) into
// per-WMMA-fragment f16 layout: frag = 32 lanes x 16 halves (32B per lane).
// B-fragment element: half h of lane L = W[k = kt*32 + (L>=16)*8 + (h&7) + (h>>3)*16][n = nt*16 + (L&15)]
// ---------------------------------------------------------------------------
__global__ void __launch_bounds__(256) prepack_kernel(
    const float* __restrict__ W1, const float* __restrict__ W2,
    f16* __restrict__ W1p, f16* __restrict__ W2p)
{
  const int W1E = K1TILES * 8 * 512;   // 53248 halves
  const int W2E = K2TILES * 8 * 512;   // 16384 halves
  int i = blockIdx.x * blockDim.x + threadIdx.x;
  if (i < W1E) {
    int frag = i >> 9, r = i & 511, lane = r >> 4, h = r & 15;
    int nt = frag / K1TILES, kt = frag % K1TILES;
    int k = kt*32 + ((lane >> 4) & 1)*8 + (h & 7) + (h >> 3)*16;
    int n = nt*16 + (lane & 15);
    W1p[i] = (k < 390) ? (f16)W1[k*128 + n] : (f16)0.f;
  } else if (i < W1E + W2E) {
    int j = i - W1E;
    int frag = j >> 9, r = j & 511, lane = r >> 4, h = r & 15;
    int nt = frag / K2TILES, kt = frag % K2TILES;
    int k = kt*32 + ((lane >> 4) & 1)*8 + (h & 7) + (h >> 3)*16;
    int n = nt*16 + (lane & 15);
    W2p[j] = (f16)W2[k*128 + n];
  }
}

// ---------------------------------------------------------------------------
// Kernel 1: fused featurization + 2 WMMA GEMM layers + VALU layer3.
// Block = 256 threads (8 waves), 128 points. Wave w owns rows [16w,16w+16)
// across all 128 output columns (8 accumulator tiles).
// ---------------------------------------------------------------------------
__global__ void __launch_bounds__(256) shade_kernel(
    const float* __restrict__ rays_o, const float* __restrict__ rays_d,
    const float* __restrict__ dgrid,  const float* __restrict__ agrid,
    const float* __restrict__ b1,     const float* __restrict__ b2,
    const float* __restrict__ W3,     const float* __restrict__ b3,
    const f16* __restrict__ W1p,      const f16* __restrict__ W2p,
    float* __restrict__ sigma_ws,     float4* __restrict__ rgb_ws)
{
  extern __shared__ f16 smem[];
  f16* feat = smem;                 // TM * FSTR halves  (layer1 A; later holds h2)
  f16* h1   = smem + TM * FSTR;     // TM * H1STR halves (layer2 A; later f32 scratch)
  const int tid = threadIdx.x;

  // ----------------- featurization: 2 threads per point, split by channel ---
  {
    const int p    = tid >> 1;
    const int part = tid & 1;
    const int gp   = blockIdx.x * TM + p;
    const int ray  = gp >> 8;
    const int s    = gp & (NSAMP - 1);
    const float dx = rays_d[ray*3+0], dy = rays_d[ray*3+1], dz = rays_d[ray*3+2];
    const float t  = 0.2f + (1.2f/255.f) * (float)s;
    float pw[3];
    pw[0] = rays_o[ray*3+0] + dx * t;
    pw[1] = rays_o[ray*3+1] + dy * t;
    pw[2] = rays_o[ray*3+2] + dz * t;
    int i0[3]; float fr[3];
    #pragma unroll
    for (int a = 0; a < 3; a++) {
      float c = (pw[a] + 1.5f) * (2.f/3.f) - 1.f;       // normalize_coord
      c = fminf(1.f, fmaxf(-1.f, c));
      float u = (c + 1.f) * 0.5f * 127.f;               // align_corners=True
      int ii = (int)floorf(u);
      ii = ii < 0 ? 0 : (ii > 126 ? 126 : ii);
      i0[a] = ii;
      fr[a] = fminf(1.f, fmaxf(0.f, u - (float)ii));
    }
    const float fx = fr[0], fy = fr[1], fz = fr[2];
    int off[8];
    off[0] = i0[2]*16384 + i0[1]*128 + i0[0];
    off[1] = off[0] + 1;      off[2] = off[0] + 128;   off[3] = off[2] + 1;
    off[4] = off[0] + 16384;  off[5] = off[4] + 1;
    off[6] = off[4] + 128;    off[7] = off[6] + 1;
    float w8[8];
    w8[0] = (1-fz)*(1-fy)*(1-fx); w8[1] = (1-fz)*(1-fy)*fx;
    w8[2] = (1-fz)*fy*(1-fx);     w8[3] = (1-fz)*fy*fx;
    w8[4] = fz*(1-fy)*(1-fx);     w8[5] = fz*(1-fy)*fx;
    w8[6] = fz*fy*(1-fx);         w8[7] = fz*fy*fx;

    f16* frow = feat + p * FSTR;
    int c_lo, c_hi;
    if (part == 0) {
      // density -> softplus -> sigma
      float sf = 0.f;
      #pragma unroll
      for (int q = 0; q < 8; q++) sf += dgrid[off[q]] * w8[q];
      sigma_ws[gp] = (sf > 15.f) ? sf : log1pf(__expf(sf));
      // view dirs + their positional encoding (cols 27..29, 354..389)
      float inv = rsqrtf(dx*dx + dy*dy + dz*dz);
      float vdn[3] = {dx*inv, dy*inv, dz*inv};
      #pragma unroll
      for (int j = 0; j < 3; j++) {
        frow[27 + j] = (f16)vdn[j];
        float fb = 1.f;
        #pragma unroll
        for (int fq = 0; fq < 6; fq++) {
          float ang = vdn[j] * fb;
          frow[354 + j*12 + fq]     = (f16)__sinf(ang);
          frow[354 + j*12 + 6 + fq] = (f16)__cosf(ang);
          fb *= 2.f;
        }
      }
      for (int k = 390; k < 416; k++) frow[k] = (f16)0.f;  // K padding
      c_lo = 0;  c_hi = 14;
    } else { c_lo = 14; c_hi = 27; }
    // appearance channels + their positional encoding (cols c, 30+12c..)
    for (int c = c_lo; c < c_hi; c++) {
      const float* g = agrid + c * CHSTRIDE;
      float av = 0.f;
      #pragma unroll
      for (int q = 0; q < 8; q++) av += g[off[q]] * w8[q];
      frow[c] = (f16)av;
      float fb = 1.f;
      #pragma unroll
      for (int fq = 0; fq < 6; fq++) {
        float ang = av * fb;
        frow[30 + c*12 + fq]     = (f16)__sinf(ang);
        frow[30 + c*12 + 6 + fq] = (f16)__cosf(ang);
        fb *= 2.f;
      }
    }
  }
  __syncthreads();

  const int lane    = tid & 31;
  const int wave    = tid >> 5;
  const int rowBase = wave * 16;
  const int nlo     = lane & 15;
  const int hi8     = (lane >> 4) & 1;
  const int mBase   = rowBase + hi8 * 8;   // D layout: vgpr r -> M = r (+8 upper lanes)

  v8f acc[8];
  const v8f zero8 = {};

  // ----------------- layer 1: [128x416]f16 x [416x128]f16 + b1, ReLU --------
  {
    const f16* aRow = feat + (rowBase + nlo) * FSTR + hi8 * 8;
    #pragma unroll
    for (int kt = 0; kt < K1TILES; kt++) {
      // batch-issue all 16 B-fragment b128 loads for this k-tile (one clause)
      v16h bf[8];
      #pragma unroll
      for (int nt = 0; nt < 8; nt++) {
        const f16* wp = W1p + (nt*K1TILES + kt)*512 + lane*16;
        bf[nt] = cat8(*(const v8h*)wp, *(const v8h*)(wp + 8));
      }
      v16h a = cat8(*(const v8h*)(aRow + kt*32),
                    *(const v8h*)(aRow + kt*32 + 16));
      #pragma unroll
      for (int nt = 0; nt < 8; nt++) {
        acc[nt] = __builtin_amdgcn_wmma_f32_16x16x32_f16(
            false, a, false, bf[nt], (short)0,
            (kt == 0) ? zero8 : acc[nt], false, false);
      }
    }
  }
  // bias + ReLU fused at store
  #pragma unroll
  for (int nt = 0; nt < 8; nt++) {
    const float bv = b1[nt*16 + nlo];
    #pragma unroll
    for (int r = 0; r < 8; r++)
      h1[(mBase + r)*H1STR + nt*16 + nlo] = (f16)fmaxf(acc[nt][r] + bv, 0.f);
  }
  __syncthreads();

  // ----------------- layer 2: [128x128] x [128x128] + b2, ReLU --------------
  {
    const f16* aRow = h1 + (rowBase + nlo) * H1STR + hi8 * 8;
    #pragma unroll
    for (int kt = 0; kt < K2TILES; kt++) {
      v16h bf[8];
      #pragma unroll
      for (int nt = 0; nt < 8; nt++) {
        const f16* wp = W2p + (nt*K2TILES + kt)*512 + lane*16;
        bf[nt] = cat8(*(const v8h*)wp, *(const v8h*)(wp + 8));
      }
      v16h a = cat8(*(const v8h*)(aRow + kt*32),
                    *(const v8h*)(aRow + kt*32 + 16));
      #pragma unroll
      for (int nt = 0; nt < 8; nt++) {
        acc[nt] = __builtin_amdgcn_wmma_f32_16x16x32_f16(
            false, a, false, bf[nt], (short)0,
            (kt == 0) ? zero8 : acc[nt], false, false);
      }
    }
  }
  // h2 (bias + ReLU) into the (now dead) feature buffer
  #pragma unroll
  for (int nt = 0; nt < 8; nt++) {
    const float bv = b2[nt*16 + nlo];
    #pragma unroll
    for (int r = 0; r < 8; r++)
      feat[(mBase + r)*FSTR + nt*16 + nlo] = (f16)fmaxf(acc[nt][r] + bv, 0.f);
  }
  __syncthreads();

  // ----------------- layer 3: 128 -> 3, sigmoid (VALU, split-K halves) ------
  {
    const int p    = tid & (TM - 1);
    const int half = tid >> 7;                 // 0 or 1
    float* pscratch = (float*)h1;              // h1 region free now (128*4 f32)
    const f16* hr = feat + p * FSTR;
    float s0 = 0.f, s1 = 0.f, s2 = 0.f;
    const int k0 = half * 64;
    for (int k = k0; k < k0 + 64; k++) {
      float hv = (float)hr[k];
      s0 = fmaf(hv, W3[k*3+0], s0);
      s1 = fmaf(hv, W3[k*3+1], s1);
      s2 = fmaf(hv, W3[k*3+2], s2);
    }
    if (half) {
      pscratch[p*4+0] = s0; pscratch[p*4+1] = s1; pscratch[p*4+2] = s2;
    }
    __syncthreads();
    if (!half) {
      s0 += pscratch[p*4+0] + b3[0];
      s1 += pscratch[p*4+1] + b3[1];
      s2 += pscratch[p*4+2] + b3[2];
      const int gp = blockIdx.x * TM + p;
      float4 o;
      o.x = 1.f / (1.f + __expf(-s0));
      o.y = 1.f / (1.f + __expf(-s1));
      o.z = 1.f / (1.f + __expf(-s2));
      o.w = 0.f;
      rgb_ws[gp] = o;                          // single global_store_b128
    }
  }
}

// ---------------------------------------------------------------------------
// Kernel 2: per-ray sequential transmittance compositing + white background.
// ---------------------------------------------------------------------------
__global__ void __launch_bounds__(256) composite_kernel(
    const float* __restrict__ rays_d,
    const float* __restrict__ sigma_ws, const float4* __restrict__ rgb_ws,
    float* __restrict__ out, int nrays)
{
  int r = blockIdx.x * blockDim.x + threadIdx.x;
  if (r >= nrays) return;
  float dx = rays_d[r*3+0], dy = rays_d[r*3+1], dz = rays_d[r*3+2];
  float dist = (1.2f/255.f) * sqrtf(dx*dx + dy*dy + dz*dz) * 25.f;
  float T = 1.f, accw = 0.f, c0 = 0.f, c1 = 0.f, c2 = 0.f;
  for (int s = 0; s < NSAMP; s++) {
    int gp = (r << 8) + s;
    float alpha = 1.f - __expf(-sigma_ws[gp] * dist);
    float w = alpha * T;
    T *= (1.f - alpha + 1e-10f);
    accw += w;
    float4 c = rgb_ws[gp];                      // one global_load_b128
    c0 += w * c.x;  c1 += w * c.y;  c2 += w * c.z;
  }
  float bg = 1.f - accw;
  out[r*3+0] = fminf(1.f, fmaxf(0.f, c0 + bg));
  out[r*3+1] = fminf(1.f, fmaxf(0.f, c1 + bg));
  out[r*3+2] = fminf(1.f, fmaxf(0.f, c2 + bg));
}

// ---------------------------------------------------------------------------
extern "C" void kernel_launch(void* const* d_in, const int* in_sizes, int n_in,
                              void* d_out, int out_size, void* d_ws, size_t ws_size,
                              hipStream_t stream) {
  (void)n_in; (void)out_size; (void)ws_size;
  const float* rays_o = (const float*)d_in[0];
  const float* rays_d = (const float*)d_in[1];
  const float* dgrid  = (const float*)d_in[2];
  const float* agrid  = (const float*)d_in[3];
  const float* W1     = (const float*)d_in[4];
  const float* b1     = (const float*)d_in[5];
  const float* W2     = (const float*)d_in[6];
  const float* b2     = (const float*)d_in[7];
  const float* W3     = (const float*)d_in[8];
  const float* b3     = (const float*)d_in[9];

  const int N     = in_sizes[0] / 3;      // 2048 rays
  const int total = N * NSAMP;            // 524288 points

  // workspace layout
  const int W1E = K1TILES * 8 * 512;      // 53248 halves
  const int W2E = K2TILES * 8 * 512;      // 16384 halves
  char* ws = (char*)d_ws;
  f16*    W1p      = (f16*)ws;
  f16*    W2p      = (f16*)(ws + (size_t)W1E * sizeof(f16));
  float*  sigma_ws = (float*)(ws + (size_t)(W1E + W2E) * sizeof(f16));
  float4* rgb_ws   = (float4*)(sigma_ws + total);
  float*  outp     = (float*)d_out;

  prepack_kernel<<<(W1E + W2E + 255)/256, 256, 0, stream>>>(W1, W2, W1p, W2p);

  const size_t smem = (size_t)(TM*FSTR + TM*H1STR) * sizeof(f16);  // 143360 B
  (void)hipFuncSetAttribute((const void*)shade_kernel,
                            hipFuncAttributeMaxDynamicSharedMemorySize, (int)smem);
  shade_kernel<<<total / TM, 256, smem, stream>>>(
      rays_o, rays_d, dgrid, agrid, b1, b2, W3, b3, W1p, W2p, sigma_ws, rgb_ws);

  composite_kernel<<<(N + 255)/256, 256, 0, stream>>>(rays_d, sigma_ws, rgb_ws, outp, N);
}
